// GraphAttentionNet_64458869178657
// MI455X (gfx1250) — compile-verified, matrix operand order checked by reading, not run on hardware
//
#include <hip/hip_runtime.h>
#include <hip/hip_bf16.h>

typedef __attribute__((ext_vector_type(16))) _Float16 v16h;
typedef __attribute__((ext_vector_type(8)))  _Float16 v8h;
typedef __attribute__((ext_vector_type(8)))  float    v8f;

#define HEADS 4
#define NEG_SLOPE 0.2f

// ---------------- utility kernels ----------------

__global__ void fill_kernel(float* __restrict__ p, float v, int n) {
    int i = blockIdx.x * blockDim.x + threadIdx.x;
    if (i < n) p[i] = v;
}

__global__ void f32_to_f16_kernel(const float* __restrict__ src, _Float16* __restrict__ dst, int n) {
    int i = blockIdx.x * blockDim.x + threadIdx.x;
    if (i < n) dst[i] = (_Float16)src[i];
}

// Pack W[K=128, Nc] (f32) into WMMA B-fragment order (f16):
//   Bp[((tileN*KC + kc)*32 + lane)*16 + i]
// lane 0-15: col = tileN*16+lane, K in {kc*32 + 0..7, +16..23};
// lanes 16-31: same cols, K in {+8..15, +24..31}. i<8 -> low group, i>=8 -> +16.
__global__ void pack_b_kernel(const float* __restrict__ W, _Float16* __restrict__ Bp, int Nc) {
    const int KC = 4;                       // 128 / 32
    int total = (Nc >> 4) * KC * 32 * 16;
    int t = blockIdx.x * blockDim.x + threadIdx.x;
    if (t >= total) return;
    int i     = t & 15;
    int lane  = (t >> 4) & 31;
    int rest  = t >> 9;                     // tileN*KC + kc
    int kc    = rest & (KC - 1);
    int tileN = rest >> 2;
    int half  = lane >> 4;
    int l15   = lane & 15;
    int k = kc * 32 + half * 8 + (i & 7) + ((i >> 3) << 4);
    int col = tileN * 16 + l15;
    Bp[t] = (_Float16)W[(size_t)k * Nc + col];
}

// ---------------- WMMA GEMM: C[M,NC] = Ah[M,128] * B[128,NC] ----------------
// Ah: f16 row-major. Bp: pre-packed f16 fragments. 5 waves/block (exact tiling
// of M=50000: 625 blocks x 5 waves x 16 rows), one 16x16 tile per wave.
// B tile (4KB) staged in LDS once per block; K=128 -> 4 v_wmma per wave.
template <int NC>
__global__ void gemm_wmma_kernel(const _Float16* __restrict__ Ah,
                                 const _Float16* __restrict__ Bp,
                                 float* __restrict__ C, int M) {
    const int K = 128, KC = 4;
    __shared__ _Float16 Bs[KC * 32 * 16];   // 2048 halves = 4KB

    const int tid = threadIdx.x;
    // cooperative B-tile load: waves 0-3 (threads 0..127) fetch 16 halves each
    if (tid < 128) {
        const _Float16* src = Bp + (size_t)blockIdx.y * (KC * 32 * 16) + tid * 16;
        *(v8h*)(Bs + tid * 16)     = *(const v8h*)(src);
        *(v8h*)(Bs + tid * 16 + 8) = *(const v8h*)(src + 8);
    }
    __syncthreads();

    const int lane  = tid & 31;
    const int wv    = tid >> 5;
    const int half  = lane >> 4;
    const int l15   = lane & 15;
    const int tileM = blockIdx.x * 5 + wv;
    const int row0  = tileM * 16;
    if (row0 >= M) return;                  // wave-uniform; M % 16 == 0

    const _Float16* ap = Ah + ((size_t)row0 + l15) * K + half * 8;
    const _Float16* bs = Bs + lane * 16;

    v8f acc = {};
#pragma unroll
    for (int kc = 0; kc < KC; ++kc) {
        union { v16h v; v8h h[2]; } a, b;
        a.h[0] = *(const v8h*)(ap + kc * 32);
        a.h[1] = *(const v8h*)(ap + kc * 32 + 16);
        b.h[0] = *(const v8h*)(bs + kc * 512);
        b.h[1] = *(const v8h*)(bs + kc * 512 + 8);
        acc = __builtin_amdgcn_wmma_f32_16x16x32_f16(
            /*neg_a=*/false, a.v, /*neg_b=*/false, b.v,
            /*c_mod=*/(short)0, acc, /*reuse_a=*/false, /*reuse_b=*/false);
    }
    // C/D layout: VGPR j -> row = row0 + 8*half + j, col = tileN*16 + l15
    float* cp = C + ((size_t)row0 + half * 8) * NC + blockIdx.y * 16 + l15;
#pragma unroll
    for (int j = 0; j < 8; ++j) cp[(size_t)j * NC] = acc[j];
}

// ---------------- attention coefficients ----------------
__global__ void attn_coef_kernel(const float* __restrict__ h,
                                 const float* __restrict__ att_src,
                                 const float* __restrict__ att_dst,
                                 float* __restrict__ as, float* __restrict__ ad,
                                 int n, int D) {
    int t = blockIdx.x * blockDim.x + threadIdx.x;
    if (t >= n * HEADS) return;
    int node = t >> 2, hd = t & 3;
    const float* hp = h + (size_t)node * HEADS * D + hd * D;
    const float* sp = att_src + hd * D;
    const float* dp = att_dst + hd * D;
    float accS = 0.f, accD = 0.f;
    for (int i = 0; i < D; ++i) { float v = hp[i]; accS += v * sp[i]; accD += v * dp[i]; }
    as[t] = accS; ad[t] = accD;
}

// ---------------- edge helpers ----------------

__device__ __forceinline__ void get_edge(const int* __restrict__ ei, int E, int e, int& s, int& d) {
    if (e < E) { s = ei[e]; d = ei[E + e]; }
    else       { s = e - E; d = s; }        // appended self-loops
}

__device__ __forceinline__ float lrelu(float x) { return x > 0.f ? x : NEG_SLOPE * x; }

__device__ __forceinline__ void atomicMaxF(float* a, float v) {
    if (v >= 0.f) atomicMax((int*)a, __float_as_int(v));
    else          atomicMin((unsigned int*)a, __float_as_uint(v));
}

__global__ void edge_max_kernel(const float* __restrict__ as, const float* __restrict__ ad,
                                const int* __restrict__ ei, int E, int nEtot,
                                float* __restrict__ m) {
    int t = blockIdx.x * blockDim.x + threadIdx.x;
    if (t >= nEtot * HEADS) return;
    int e = t >> 2, hd = t & 3, s, d;
    get_edge(ei, E, e, s, d);
    float x = lrelu(as[s * HEADS + hd] + ad[d * HEADS + hd]);
    atomicMaxF(&m[d * HEADS + hd], x);
}

__global__ void edge_sum_kernel(const float* __restrict__ as, const float* __restrict__ ad,
                                const float* __restrict__ m,
                                const int* __restrict__ ei, int E, int nEtot,
                                float* __restrict__ den) {
    int t = blockIdx.x * blockDim.x + threadIdx.x;
    if (t >= nEtot * HEADS) return;
    int e = t >> 2, hd = t & 3, s, d;
    get_edge(ei, E, e, s, d);
    float x = lrelu(as[s * HEADS + hd] + ad[d * HEADS + hd]);
    atomicAdd(&den[d * HEADS + hd], __expf(x - m[d * HEADS + hd]));
}

// out[dst, c] += alpha(e, head(c)) * h[src, c]; one thread per (edge, channel)
__global__ void edge_agg_kernel(const float* __restrict__ as, const float* __restrict__ ad,
                                const float* __restrict__ m, const float* __restrict__ den,
                                const float* __restrict__ hsrc,
                                const int* __restrict__ ei, int E, int nEtot,
                                int logHD, int logD,
                                float* __restrict__ out) {
    int t = blockIdx.x * blockDim.x + threadIdx.x;
    if (t >= (nEtot << logHD)) return;
    int e  = t >> logHD;
    int c  = t & ((1 << logHD) - 1);
    int hd = c >> logD;
    int s, d;
    get_edge(ei, E, e, s, d);
    float x     = lrelu(as[s * HEADS + hd] + ad[d * HEADS + hd]);
    float alpha = __expf(x - m[d * HEADS + hd]) / (den[d * HEADS + hd] + 1e-16f);
    atomicAdd(&out[((size_t)d << logHD) + c], alpha * hsrc[((size_t)s << logHD) + c]);
}

// z16 = (f16) relu(agg + b)   (feeds next WMMA GEMM directly)
__global__ void bias_relu_f16_kernel(const float* __restrict__ agg, const float* __restrict__ b,
                                     _Float16* __restrict__ z, int n, int C) {
    int t = blockIdx.x * blockDim.x + threadIdx.x;
    if (t >= n * C) return;
    float v = agg[t] + b[t & (C - 1)];      // C is a power of two (128)
    z[t] = (_Float16)(v > 0.f ? v : 0.f);
}

// final: h = agg + b2 -> out[0:N*C); log_softmax(h) -> out[N*C : 2*N*C)
__global__ void finalize_kernel(const float* __restrict__ agg, const float* __restrict__ b,
                                float* __restrict__ out, int n, int C) {
    int node = blockIdx.x * blockDim.x + threadIdx.x;
    if (node >= n) return;
    const float* r = agg + (size_t)node * C;
    float* hrow = out + (size_t)node * C;
    float* lrow = out + (size_t)n * C + (size_t)node * C;
    float mx = -3.4e38f;
    for (int c = 0; c < C; ++c) { float v = r[c] + b[c]; hrow[c] = v; mx = fmaxf(mx, v); }
    float s = 0.f;
    for (int c = 0; c < C; ++c) s += __expf(hrow[c] - mx);
    float lse = mx + __logf(s);
    for (int c = 0; c < C; ++c) lrow[c] = hrow[c] - lse;
}

// ---------------- driver ----------------

static inline int nblk(long long n, int b) { return (int)((n + b - 1) / b); }

extern "C" void kernel_launch(void* const* d_in, const int* in_sizes, int n_in,
                              void* d_out, int out_size, void* d_ws, size_t ws_size,
                              hipStream_t stream) {
    const float* point = (const float*)d_in[0];
    const int*   ei    = (const int*)d_in[1];     // [2,E]: src then dst
    const float* W1    = (const float*)d_in[2];
    const float* b1    = (const float*)d_in[3];
    const float* asr1  = (const float*)d_in[4];
    const float* ads1  = (const float*)d_in[5];
    const float* W2    = (const float*)d_in[6];
    const float* b2    = (const float*)d_in[7];
    const float* asr2  = (const float*)d_in[8];
    const float* ads2  = (const float*)d_in[9];

    const int DIN = 128, D1 = 32, D2 = 16;
    const int HD1 = HEADS * D1;          // 128
    const int HD2 = HEADS * D2;          // 64
    const int Nn    = in_sizes[0] / DIN; // 50000
    const int E     = in_sizes[1] / 2;   // 800000
    const int nEtot = E + Nn;

    // workspace layout (float units)
    float* ws = (float*)d_ws;
    size_t off = 0;
    float* h1   = ws + off; off += (size_t)Nn * HD1;
    float* o1   = ws + off; off += (size_t)Nn * HD1;
    float* h2   = ws + off; off += (size_t)Nn * HD2;
    float* o2   = ws + off; off += (size_t)Nn * HD2;
    float* as1  = ws + off; off += (size_t)Nn * HEADS;
    float* ad1  = ws + off; off += (size_t)Nn * HEADS;
    float* m1   = ws + off; off += (size_t)Nn * HEADS;
    float* den1 = ws + off; off += (size_t)Nn * HEADS;
    float* as2  = ws + off; off += (size_t)Nn * HEADS;
    float* ad2  = ws + off; off += (size_t)Nn * HEADS;
    float* m2   = ws + off; off += (size_t)Nn * HEADS;
    float* den2 = ws + off; off += (size_t)Nn * HEADS;
    _Float16* Xh  = (_Float16*)(ws + off); off += ((size_t)Nn * DIN) / 2;  // x as f16
    _Float16* Z1h = (_Float16*)(ws + off); off += ((size_t)Nn * HD1) / 2;  // relu(l1) f16
    _Float16* B1p = (_Float16*)(ws + off); off += (DIN * HD1) / 2;         // packed W1
    _Float16* B2p = (_Float16*)(ws + off); off += (HD1 * HD2) / 2;         // packed W2

    const int B = 256;
    const int Mtiles = (Nn + 15) / 16;   // 3125

    // re-init accumulators every call (harness does not re-poison between replays)
    fill_kernel<<<nblk((long long)Nn * HD1, B), B, 0, stream>>>(o1, 0.f, Nn * HD1);
    fill_kernel<<<nblk((long long)Nn * HD2, B), B, 0, stream>>>(o2, 0.f, Nn * HD2);
    fill_kernel<<<nblk((long long)Nn * HEADS, B), B, 0, stream>>>(den1, 0.f, Nn * HEADS);
    fill_kernel<<<nblk((long long)Nn * HEADS, B), B, 0, stream>>>(den2, 0.f, Nn * HEADS);
    fill_kernel<<<nblk((long long)Nn * HEADS, B), B, 0, stream>>>(m1, -3.0e38f, Nn * HEADS);
    fill_kernel<<<nblk((long long)Nn * HEADS, B), B, 0, stream>>>(m2, -3.0e38f, Nn * HEADS);

    // operand preparation: f16 A copy of x, fragment-packed f16 weights
    f32_to_f16_kernel<<<nblk((long long)Nn * DIN, B), B, 0, stream>>>(point, Xh, Nn * DIN);
    pack_b_kernel<<<nblk(DIN * HD1, B), B, 0, stream>>>(W1, B1p, HD1);
    pack_b_kernel<<<nblk(HD1 * HD2, B), B, 0, stream>>>(W2, B2p, HD2);

    // ---- layer 1 ----
    {
        dim3 g((Mtiles + 4) / 5, HD1 / 16);
        gemm_wmma_kernel<128><<<g, 160, 0, stream>>>(Xh, B1p, h1, Nn);
    }
    attn_coef_kernel<<<nblk((long long)Nn * HEADS, B), B, 0, stream>>>(h1, asr1, ads1, as1, ad1, Nn, D1);
    edge_max_kernel<<<nblk((long long)nEtot * HEADS, B), B, 0, stream>>>(as1, ad1, ei, E, nEtot, m1);
    edge_sum_kernel<<<nblk((long long)nEtot * HEADS, B), B, 0, stream>>>(as1, ad1, m1, ei, E, nEtot, den1);
    edge_agg_kernel<<<nblk((long long)nEtot * HD1, B), B, 0, stream>>>(as1, ad1, m1, den1, h1, ei, E, nEtot, 7, 5, o1);
    bias_relu_f16_kernel<<<nblk((long long)Nn * HD1, B), B, 0, stream>>>(o1, b1, Z1h, Nn, HD1);

    // ---- layer 2 ----
    {
        dim3 g((Mtiles + 4) / 5, HD2 / 16);
        gemm_wmma_kernel<64><<<g, 160, 0, stream>>>(Z1h, B2p, h2, Nn);
    }
    attn_coef_kernel<<<nblk((long long)Nn * HEADS, B), B, 0, stream>>>(h2, asr2, ads2, as2, ad2, Nn, D2);
    edge_max_kernel<<<nblk((long long)nEtot * HEADS, B), B, 0, stream>>>(as2, ad2, ei, E, nEtot, m2);
    edge_sum_kernel<<<nblk((long long)nEtot * HEADS, B), B, 0, stream>>>(as2, ad2, m2, ei, E, nEtot, den2);
    edge_agg_kernel<<<nblk((long long)nEtot * HD2, B), B, 0, stream>>>(as2, ad2, m2, den2, h2, ei, E, nEtot, 6, 4, o2);

    // ---- output: h and log_softmax(h) ----
    finalize_kernel<<<nblk(Nn, B), B, 0, stream>>>(o2, b2, (float*)d_out, Nn, HD2);
}
